// PTGAMiniExpert_5858335392199
// MI455X (gfx1250) — compile-verified, matrix operand order checked by reading, not run on hardware
//
#include <hip/hip_runtime.h>
#include <hip/hip_bf16.h>

#define BGRAPH 1024
#define NNODE  64
#define DIN    64
#define HID    128
#define HEADS  4
#define EPG    512                  // edges per graph (448 random + 64 self loops)
#define ETOT   (BGRAPH * EPG)
#define LN_EPS 1e-5f
#define NEG_SLOPE 0.2f

typedef __attribute__((ext_vector_type(16))) __bf16 v16bf;
typedef __attribute__((ext_vector_type(8)))  float  v8f;

union FragBF { v16bf v; float4 f4[2]; };

// Load a 16x32 bf16 WMMA fragment from a row-major [rows][ld] bf16 array.
// CDNA5 16-bit A layout: lanes 0-15 -> K {k0..k0+7, k0+16..k0+23},
// lanes 16-31 -> K {k0+8..k0+15, k0+24..k0+31}. B (from transposed weights)
// mirrors this, so the same loader works with the column index as "row".
__device__ __forceinline__ v16bf load_frag(const __bf16* base, int tile_row, int ld, int k0, int lane) {
    int half = lane >> 4;
    int r    = lane & 15;
    const __bf16* p = base + (size_t)(tile_row + r) * ld + k0 + half * 8;
    FragBF f;
    f.f4[0] = *(const float4*)(p);
    f.f4[1] = *(const float4*)(p + 16);
    return f.v;
}

__device__ __forceinline__ v8f wmma_bf16(v16bf a, v16bf b, v8f c) {
    return __builtin_amdgcn_wmma_f32_16x16x32_bf16(false, a, false, b, (short)0, c, false, false);
}

// 64 x K x 128 GEMM slice for one wave: ni = wave owns one 16-wide N tile,
// 4 M tiles accumulated in registers; B fragment loaded once per k-step and
// reused by all 4 WMMAs (back-to-back WMMA operand reuse).
__device__ __forceinline__ void gemm_wave(const __bf16* A, int ldA,
                                          const __bf16* Bt, int ldB,
                                          int K, int ni, int lane, v8f acc[4]) {
    for (int k0 = 0; k0 < K; k0 += 32) {
        v16bf b = load_frag(Bt, ni * 16, ldB, k0, lane);
        #pragma unroll
        for (int mi = 0; mi < 4; mi++) {
            v16bf a = load_frag(A, mi * 16, ldA, k0, lane);
            acc[mi] = wmma_bf16(a, b, acc[mi]);
        }
    }
}

// gfx1250 async global->LDS copy (ASYNCcnt-tracked), 16B per lane.
__device__ __forceinline__ void async_copy_b128(unsigned lds_off, const void* gaddr) {
    asm volatile("global_load_async_to_lds_b128 %0, %1, off"
                 :: "v"(lds_off), "v"((unsigned long long)(uintptr_t)gaddr)
                 : "memory");
}
__device__ __forceinline__ void wait_async0() {
    asm volatile("s_wait_asynccnt 0x0" ::: "memory");
}

// ---- prep: transpose + bf16-cast the three weight matrices into workspace ----
__global__ void ptga_prep_weights(const float* __restrict__ W_align,
                                  const float* __restrict__ W1,
                                  const float* __restrict__ W2,
                                  __bf16* __restrict__ WAt,   // [HID][DIN]
                                  __bf16* __restrict__ W1t,   // [HEADS*HID][HID]
                                  __bf16* __restrict__ W2t) { // [HID][HEADS*HID]
    int i = blockIdx.x * blockDim.x + threadIdx.x;
    if (i < DIN * HID) {                 // WAt[n][k] = W_align[k][n], ld=DIN
        int n = i / DIN, k = i % DIN;
        WAt[i] = (__bf16)W_align[k * HID + n];
    }
    if (i < HID * HEADS * HID) {         // W1t[n][k] = W1[k][n], ld=HID
        int n = i / HID, k = i % HID;
        W1t[i] = (__bf16)W1[k * (HEADS * HID) + n];
    }
    if (i < (HEADS * HID) * HID) {       // W2t[n][k] = W2[k][n], ld=HEADS*HID
        int n = i / (HEADS * HID), k = i % (HEADS * HID);
        W2t[i] = (__bf16)W2[k * HID + n];
    }
}

// ---- main: one workgroup (8 waves) per graph ----
__global__ __launch_bounds__(256)
void ptga_graph_kernel(const float* __restrict__ fv,
                       const float* __restrict__ b_align,
                       const float* __restrict__ ln_gamma,
                       const float* __restrict__ ln_beta,
                       const float* __restrict__ mask_logits,
                       const float* __restrict__ att_src1,
                       const float* __restrict__ att_dst1,
                       const float* __restrict__ bias1,
                       const float* __restrict__ att_src2,
                       const float* __restrict__ att_dst2,
                       const float* __restrict__ bias2,
                       const int*   __restrict__ edge_index,
                       const __bf16* __restrict__ WAt,
                       const __bf16* __restrict__ W1t,
                       const __bf16* __restrict__ W2t,
                       float* __restrict__ out) {
    __shared__ __align__(16) __bf16 sh_x2[NNODE * HEADS * HID]; // 64KB: GAT1 out (bf16) / GAT2 out (f32 reuse)
    __shared__ __align__(16) __bf16 sh_h1[NNODE * HID];         // 16KB: per-head h1 (bf16)
    __shared__ __align__(16) float  sh_xf[NNODE * HID];         // 32KB: fv staging / x (f32) / h2 (f32)
    __shared__ __align__(16) __bf16 sh_xb[NNODE * HID];         // 16KB: gated x (bf16) = GAT1 A matrix
    __shared__ __align__(16) __bf16 sh_fv[NNODE * DIN];         //  8KB: field vecs (bf16)
    __shared__ int   sh_src[EPG];
    __shared__ int   sh_dst[EPG];
    __shared__ float sh_sc[EPG];                                // scores -> alpha, in place
    __shared__ float sh_red[NNODE * 4 * 2];                     // partial reductions
    __shared__ float sh_av[NNODE], sh_ad[NNODE];                // a_src / a_dst per node
    __shared__ int   sh_cnt[NNODE];                             // CSR: in-degree
    __shared__ int   sh_off[NNODE];                             // CSR: offsets
    __shared__ int   sh_fill[NNODE];                            // CSR: fill cursors
    __shared__ unsigned short sh_csr[EPG];                      // CSR: edge ids by dst

    const int t    = threadIdx.x;
    const int wave = t >> 5;
    const int lane = t & 31;
    const int g    = blockIdx.x;

    // ---------- phase 0: async-stage fv, load edges, build per-dst CSR ----------
    {   // fv f32 -> sh_xf staging via gfx1250 async LDS loads (ASYNCcnt path)
        const float* fvg = fv + (size_t)g * NNODE * DIN;
        unsigned lds_base = (unsigned)(uintptr_t)(void*)sh_xf;
        for (int i = t; i < (NNODE * DIN) / 4; i += 256)
            async_copy_b128(lds_base + (unsigned)i * 16u, (const void*)(((const float4*)fvg) + i));
    }
    for (int e = t; e < EPG; e += 256) {
        sh_src[e] = edge_index[(size_t)g * EPG + e] - g * NNODE;
        sh_dst[e] = edge_index[(size_t)ETOT + (size_t)g * EPG + e] - g * NNODE;
    }
    if (t < NNODE) { sh_cnt[t] = 0; sh_fill[t] = 0; }
    wait_async0();
    __syncthreads();
    // convert staged fv to bf16
    for (int i = t; i < NNODE * DIN; i += 256) sh_fv[i] = (__bf16)sh_xf[i];
    // CSR pass 1: in-degrees
    for (int e = t; e < EPG; e += 256) atomicAdd(&sh_cnt[sh_dst[e]], 1);
    __syncthreads();
    if (t == 0) {                     // tiny serial prefix sum over 64 bins
        int run = 0;
        for (int d = 0; d < NNODE; d++) { sh_off[d] = run; run += sh_cnt[d]; }
    }
    __syncthreads();
    // CSR pass 2: scatter edge ids
    for (int e = t; e < EPG; e += 256) {
        int d = sh_dst[e];
        int p = atomicAdd(&sh_fill[d], 1);
        sh_csr[sh_off[d] + p] = (unsigned short)e;
    }
    __syncthreads();

    // ---------- phase 1: align GEMM  x = fv @ W_align  (64x64x128) ----------
    {
        v8f acc[4] = {};
        gemm_wave(sh_fv, DIN, WAt, DIN, DIN, wave, lane, acc);
        int r0 = (lane >> 4) * 8;
        int c  = wave * 16 + (lane & 15);
        #pragma unroll
        for (int mi = 0; mi < 4; mi++)
            #pragma unroll
            for (int i = 0; i < 8; i++)
                sh_xf[(mi * 16 + r0 + i) * HID + c] = acc[mi][i];
    }
    __syncthreads();

    // ---------- phase 2: bias + LayerNorm + sigmoid gate -> bf16 ----------
    {
        int row = t >> 2, q = t & 3;
        float s0 = 0.f, s1 = 0.f;
        #pragma unroll 8
        for (int j = 0; j < 32; j++) {
            int c = q * 32 + j;
            float v = sh_xf[row * HID + c] + b_align[c];
            s0 += v; s1 += v * v;
        }
        sh_red[(row * 4 + q) * 2 + 0] = s0;
        sh_red[(row * 4 + q) * 2 + 1] = s1;
        __syncthreads();
        float sum = 0.f, sq = 0.f;
        #pragma unroll
        for (int p = 0; p < 4; p++) {
            sum += sh_red[(row * 4 + p) * 2 + 0];
            sq  += sh_red[(row * 4 + p) * 2 + 1];
        }
        float mu  = sum * (1.0f / HID);
        float var = sq * (1.0f / HID) - mu * mu;
        float inv = rsqrtf(var + LN_EPS);
        float gate = 1.0f / (1.0f + __expf(-mask_logits[row]));
        #pragma unroll 8
        for (int j = 0; j < 32; j++) {
            int c = q * 32 + j;
            float v = sh_xf[row * HID + c] + b_align[c];
            v = (v - mu) * inv * ln_gamma[c] + ln_beta[c];
            sh_xb[row * HID + c] = (__bf16)(v * gate);
        }
    }
    __syncthreads();

    // ---------- phase 3: GAT layer 1, head by head ----------
    for (int h = 0; h < HEADS; h++) {
        // 3a: h1_h = x @ W1[:, h*128 : (h+1)*128]   (64x128x128)
        {
            v8f acc[4] = {};
            gemm_wave(sh_xb, HID, W1t + (size_t)(h * HID) * HID, HID, HID, wave, lane, acc);
            int r0 = (lane >> 4) * 8;
            int c  = wave * 16 + (lane & 15);
            #pragma unroll
            for (int mi = 0; mi < 4; mi++)
                #pragma unroll
                for (int i = 0; i < 8; i++)
                    sh_h1[(mi * 16 + r0 + i) * HID + c] = (__bf16)acc[mi][i];
        }
        __syncthreads();

        // 3b: attention dot products a_src / a_dst per node
        {
            int row = t >> 2, q = t & 3;
            float s0 = 0.f, s1 = 0.f;
            #pragma unroll 8
            for (int j = 0; j < 32; j++) {
                int c = q * 32 + j;
                float hv = (float)sh_h1[row * HID + c];
                s0 += hv * att_src1[h * HID + c];
                s1 += hv * att_dst1[h * HID + c];
            }
            sh_red[(row * 4 + q) * 2 + 0] = s0;
            sh_red[(row * 4 + q) * 2 + 1] = s1;
        }
        __syncthreads();
        {
            int row = t >> 2, q = t & 3;
            if (q == 0) {
                float a = 0.f, b = 0.f;
                #pragma unroll
                for (int p = 0; p < 4; p++) {
                    a += sh_red[(row * 4 + p) * 2 + 0];
                    b += sh_red[(row * 4 + p) * 2 + 1];
                }
                sh_av[row] = a; sh_ad[row] = b;
            }
        }
        __syncthreads();

        // 3c: per-dst segment softmax using the CSR (one node per thread)
        if (t < NNODE) {
            int o = sh_off[t], n = sh_cnt[t];
            float ad = sh_ad[t];
            float m = -1e30f;
            for (int i = 0; i < n; i++) {
                int e = sh_csr[o + i];
                float sc = sh_av[sh_src[e]] + ad;
                sc = sc > 0.f ? sc : sc * NEG_SLOPE;
                sh_sc[e] = sc;
                m = fmaxf(m, sc);
            }
            float s = 0.f;
            for (int i = 0; i < n; i++) {
                int e = sh_csr[o + i];
                float v = __expf(sh_sc[e] - m);
                sh_sc[e] = v;
                s += v;
            }
            float inv = 1.0f / (s + 1e-16f);
            for (int i = 0; i < n; i++) sh_sc[sh_csr[o + i]] *= inv;
        }
        __syncthreads();

        // 3d: message aggregation + bias1 + ELU -> sh_x2 columns of this head
        {
            int d = t >> 2, q = t & 3;
            int o = sh_off[d], n = sh_cnt[d];
            float acc[32];
            #pragma unroll
            for (int j = 0; j < 32; j++) acc[j] = 0.f;
            for (int i = 0; i < n; i++) {
                int e = sh_csr[o + i];
                float a = sh_sc[e];
                const __bf16* hp = sh_h1 + sh_src[e] * HID + q * 32;
                #pragma unroll
                for (int j = 0; j < 32; j++) acc[j] += a * (float)hp[j];
            }
            int cbase = h * HID + q * 32;
            #pragma unroll
            for (int j = 0; j < 32; j++) {
                float v = acc[j] + bias1[cbase + j];
                v = v > 0.f ? v : (__expf(v) - 1.0f);   // ELU
                sh_x2[d * (HEADS * HID) + cbase + j] = (__bf16)v;
            }
        }
        __syncthreads();
    }

    // ---------- phase 4: GAT2 GEMM  h2 = x2 @ W2  (64x512x128) ----------
    {
        v8f acc[4] = {};
        gemm_wave(sh_x2, HEADS * HID, W2t, HEADS * HID, HEADS * HID, wave, lane, acc);
        int r0 = (lane >> 4) * 8;
        int c  = wave * 16 + (lane & 15);
        #pragma unroll
        for (int mi = 0; mi < 4; mi++)
            #pragma unroll
            for (int i = 0; i < 8; i++)
                sh_xf[(mi * 16 + r0 + i) * HID + c] = acc[mi][i];   // h2 f32
    }
    __syncthreads();

    // ---------- phase 5: attention for layer 2 (1 head) ----------
    {
        int row = t >> 2, q = t & 3;
        float s0 = 0.f, s1 = 0.f;
        #pragma unroll 8
        for (int j = 0; j < 32; j++) {
            int c = q * 32 + j;
            float hv = sh_xf[row * HID + c];
            s0 += hv * att_src2[c];
            s1 += hv * att_dst2[c];
        }
        sh_red[(row * 4 + q) * 2 + 0] = s0;
        sh_red[(row * 4 + q) * 2 + 1] = s1;
    }
    __syncthreads();
    {
        int row = t >> 2, q = t & 3;
        if (q == 0) {
            float a = 0.f, b = 0.f;
            #pragma unroll
            for (int p = 0; p < 4; p++) {
                a += sh_red[(row * 4 + p) * 2 + 0];
                b += sh_red[(row * 4 + p) * 2 + 1];
            }
            sh_av[row] = a; sh_ad[row] = b;
        }
    }
    __syncthreads();
    if (t < NNODE) {
        int o = sh_off[t], n = sh_cnt[t];
        float ad = sh_ad[t];
        float m = -1e30f;
        for (int i = 0; i < n; i++) {
            int e = sh_csr[o + i];
            float sc = sh_av[sh_src[e]] + ad;
            sc = sc > 0.f ? sc : sc * NEG_SLOPE;
            sh_sc[e] = sc;
            m = fmaxf(m, sc);
        }
        float s = 0.f;
        for (int i = 0; i < n; i++) {
            int e = sh_csr[o + i];
            float v = __expf(sh_sc[e] - m);
            sh_sc[e] = v;
            s += v;
        }
        float inv = 1.0f / (s + 1e-16f);
        for (int i = 0; i < n; i++) sh_sc[sh_csr[o + i]] *= inv;
    }
    __syncthreads();

    // ---------- phase 6: aggregation 2 + bias2, then mean pool ----------
    float* sh_out = (float*)sh_x2;   // reuse (x2 is dead after GEMM2)
    {
        int d = t >> 2, q = t & 3;
        int o = sh_off[d], n = sh_cnt[d];
        float acc[32];
        #pragma unroll
        for (int j = 0; j < 32; j++) acc[j] = 0.f;
        for (int i = 0; i < n; i++) {
            int e = sh_csr[o + i];
            float a = sh_sc[e];
            const float* hp = sh_xf + sh_src[e] * HID + q * 32;
            #pragma unroll
            for (int j = 0; j < 32; j++) acc[j] += a * hp[j];
        }
        #pragma unroll
        for (int j = 0; j < 32; j++) {
            int c = q * 32 + j;
            sh_out[d * HID + c] = acc[j] + bias2[c];
        }
    }
    __syncthreads();
    if (t < HID) {
        float s = 0.f;
        for (int r = 0; r < NNODE; r++) s += sh_out[r * HID + t];
        out[(size_t)g * HID + t] = s * (1.0f / NNODE);
    }
    if (g == 0 && t < NNODE) {
        out[(size_t)BGRAPH * HID + t] = 1.0f / (1.0f + __expf(-mask_logits[t]));
    }
}

extern "C" void kernel_launch(void* const* d_in, const int* in_sizes, int n_in,
                              void* d_out, int out_size, void* d_ws, size_t ws_size,
                              hipStream_t stream) {
    const float* fv          = (const float*)d_in[0];
    const float* W_align     = (const float*)d_in[1];
    const float* b_align     = (const float*)d_in[2];
    const float* ln_gamma    = (const float*)d_in[3];
    const float* ln_beta     = (const float*)d_in[4];
    const float* mask_logits = (const float*)d_in[5];
    const float* W1          = (const float*)d_in[6];
    const float* att_src1    = (const float*)d_in[7];
    const float* att_dst1    = (const float*)d_in[8];
    const float* bias1       = (const float*)d_in[9];
    const float* W2          = (const float*)d_in[10];
    const float* att_src2    = (const float*)d_in[11];
    const float* att_dst2    = (const float*)d_in[12];
    const float* bias2       = (const float*)d_in[13];
    const int*   edge_index  = (const int*)d_in[14];

    __bf16* WAt = (__bf16*)d_ws;                       // [128][64]
    __bf16* W1t = WAt + DIN * HID;                     // [512][128]
    __bf16* W2t = W1t + HID * HEADS * HID;             // [128][512]

    ptga_prep_weights<<<dim3(256), dim3(256), 0, stream>>>(W_align, W1, W2, WAt, W1t, W2t);
    ptga_graph_kernel<<<dim3(BGRAPH), dim3(256), 0, stream>>>(
        fv, b_align, ln_gamma, ln_beta, mask_logits,
        att_src1, att_dst1, bias1, att_src2, att_dst2, bias2,
        edge_index, WAt, W1t, W2t, (float*)d_out);
}
